// AxialAttentionModule_85315230368173
// MI455X (gfx1250) — matrix operand order repaired and checked by reading, not compiled, not run on hardware
//
#include <hip/hip_runtime.h>

// ---------------------------------------------------------------------------
// Types and WMMA helpers (gfx1250 / CDNA5, wave32)
// ---------------------------------------------------------------------------
typedef unsigned short bf16_t;
typedef __attribute__((ext_vector_type(16))) __bf16 v16bf;
typedef __attribute__((ext_vector_type(2)))  __bf16 v2bf;
typedef __attribute__((ext_vector_type(8)))  float  v8f;

union ABTile  { v16bf v; uint4 q[2]; };
union AccTile { v8f v; float f[8]; };

#if defined(__has_builtin)
#  if __has_builtin(__builtin_amdgcn_cvt_pk_bf16_f32)
#    define HAVE_CVT_PK_BF16 1
#  endif
#endif
#ifndef HAVE_CVT_PK_BF16
#  define HAVE_CVT_PK_BF16 0
#endif

__device__ __forceinline__ unsigned pack2bf(float a, float b) {
#if HAVE_CVT_PK_BF16
  union { v2bf v; unsigned u; } u;
  u.v = __builtin_amdgcn_cvt_pk_bf16_f32(a, b);
  return u.u;
#else
  union { float f; unsigned u; } va, vb;
  va.f = a; vb.f = b;
  unsigned ra = va.u + 0x7FFFu + ((va.u >> 16) & 1u);
  unsigned rb = vb.u + 0x7FFFu + ((vb.u >> 16) & 1u);
  return (ra >> 16) | (rb & 0xFFFF0000u);
#endif
}
__device__ __forceinline__ bf16_t f2bf(float x) {
#if HAVE_CVT_PK_BF16
  return (bf16_t)(pack2bf(x, 0.f) & 0xFFFFu);
#else
  union { float f; unsigned u; } v; v.f = x;
  unsigned r = v.u + 0x7FFFu + ((v.u >> 16) & 1u);
  return (bf16_t)(r >> 16);
#endif
}

// Async global->LDS copy (CDNA5, ASYNCcnt tracked). lds_byte_off is the byte
// offset inside the workgroup LDS allocation (dynamic LDS starts at 0 here).
__device__ __forceinline__ void async_ld_b128(unsigned lds_byte_off, const void* g) {
  asm volatile("global_load_async_to_lds_b128 %0, %1, off"
               :: "v"(lds_byte_off), "v"((unsigned long long)(uintptr_t)g)
               : "memory");
}
__device__ __forceinline__ void wait_async0() {
  asm volatile("s_wait_asynccnt 0" ::: "memory");
}

// A-operand (16x32, 16-bit): lane&15 = M row; lanes 0-15 hold K 0-7 & 16-23,
// lanes 16-31 hold K 8-15 & 24-31. Source buffer is row-major [M][K] bf16.
__device__ __forceinline__ v16bf load_a_tile(const bf16_t* base, int pitch, int lane) {
  ABTile t;
  const char* p = (const char*)(base + (lane & 15) * pitch) + ((lane >> 4) << 4);
  t.q[0] = *(const uint4*)(p);
  t.q[1] = *(const uint4*)(p + 32);
  return t.v;
}
// B-operand (32x16, 16-bit): lane&15 = N col; lanes 0-15 hold K 0-7 & 16-23,
// lanes 16-31 hold K 8-15 & 24-31. Source buffer is row-major [N][K] bf16 (B^T).
__device__ __forceinline__ v16bf load_b_tile(const bf16_t* base, int pitch, int lane) {
  ABTile t;
  const uint4* p = (const uint4*)(base + (lane & 15) * pitch + ((lane >> 4) << 3));
  t.q[0] = p[0];
  t.q[1] = p[2];
  return t.v;
}
__device__ __forceinline__ v8f wmma_bf16(v16bf a, v16bf b, v8f c) {
  return __builtin_amdgcn_wmma_f32_16x16x32_bf16(false, a, false, b, (short)0, c,
                                                 false, false);
}

// ---------------------------------------------------------------------------
// Problem constants
// ---------------------------------------------------------------------------
#define BATCH 4
#define CCH   256
#define LSEQ  128
#define NHEAD 8
#define DH    32
#define NTHR  512          // 16 waves per workgroup
#define NWAVE 16

// LDS layout (bytes). Pitches in bf16 elements give 16B-aligned fragment rows
// that shift 4 LDS banks per row (conflict-free b128 tile loads).
#define PX 264   // [l][c] buffers (X^T, Q^T, K^T, Out^T)  row = 528 B
#define PVV 136  // V [d][l]                               row = 272 B
#define PP 40    // per-wave P scratch [16][40]            row = 80 B
#define PF 260   // fused staging f32 [w][c]               row = 1040 B
#define OFF_X 0                   // 128*264*2 = 67584   (also sOut, phase 5+)
#define OFF_Q 67584               // 67584
#define OFF_K 135168              // 67584
#define OFF_V 202752              // 256*136*2 = 69632
#define OFF_P 272384              // 16*16*40*2 = 20480 (per-wave P)
#define OFF_B 292864              // 8*256*4    = 8192
#define SMEM_ATT 301056           // total < 320 KB
// mode-1 fused staging (f32, 133120 B) aliases sQ+sK (135168 B)
// mode-0 output staging (bf16, 67584 B) aliases sQ

// ---------------------------------------------------------------------------
// Weight pack: W[c][d] f32 -> WT[d][c] bf16  (A/B operand ready, row-major)
// ---------------------------------------------------------------------------
__global__ void pack_w_kernel(const float* __restrict__ W, bf16_t* __restrict__ out) {
  int idx = blockIdx.x * 256 + threadIdx.x;   // 65536 total
  int d = idx >> 8, c = idx & 255;
  out[idx] = f2bf(W[c * 256 + d]);
}

// ---------------------------------------------------------------------------
// Prep: BN + channel-last transposes + ref passthrough.
//   Tq_row [b][w][h][c] = bn(tgt)      (row-attn layouts)
//   Tkv_row[b][w][h][c] = bn(ref)
//   Tkv_col[b][h][w][c] = ref (bf16)
//   ref_out = ref (f32, second tuple output)
// ---------------------------------------------------------------------------
__global__ __launch_bounds__(256, 4) void prep_kernel(
    const float* __restrict__ tgt, const float* __restrict__ ref,
    const float* __restrict__ tg, const float* __restrict__ tb,
    const float* __restrict__ tm, const float* __restrict__ tv,
    const float* __restrict__ rg, const float* __restrict__ rb,
    const float* __restrict__ rm, const float* __restrict__ rv,
    bf16_t* __restrict__ Tq_row, bf16_t* __restrict__ Tkv_row,
    bf16_t* __restrict__ Tkv_col, float* __restrict__ ref_out)
{
  __shared__ bf16_t sT[64][68], sR[64][68], sW[64][68];
  const int b = blockIdx.z, hh = blockIdx.y;
  const int ct0 = (blockIdx.x & 3) << 6, wt0 = (blockIdx.x >> 2) << 6;
  const int tid = threadIdx.x;
  const int wi_r = tid & 63, cq = tid >> 6;
  #pragma unroll
  for (int i = 0; i < 16; ++i) {
    int ci = (i << 2) + cq;
    int c = ct0 + ci, w = wt0 + wi_r;
    size_t gi = (((size_t)b * CCH + c) * LSEQ + hh) * LSEQ + w;
    float xt = tgt[gi], xr = ref[gi];
    float ti = tg[c] * rsqrtf(tv[c] + 1e-5f);
    float ri = rg[c] * rsqrtf(rv[c] + 1e-5f);
    sT[wi_r][ci] = f2bf(xt * ti + (tb[c] - tm[c] * ti));
    sR[wi_r][ci] = f2bf(xr * ri + (rb[c] - rm[c] * ri));
    sW[wi_r][ci] = f2bf(xr);
    ref_out[gi] = xr;
  }
  __syncthreads();
  const int wi = tid >> 2, c0 = (tid & 3) << 4;
  const int w = wt0 + wi;
  size_t orow = (((size_t)b * LSEQ + w) * LSEQ + hh) * CCH + ct0 + c0;
  size_t ocol = (((size_t)b * LSEQ + hh) * LSEQ + w) * CCH + ct0 + c0;
  unsigned* dq = (unsigned*)(Tq_row + orow);
  unsigned* dr = (unsigned*)(Tkv_row + orow);
  unsigned* dc = (unsigned*)(Tkv_col + ocol);
  #pragma unroll
  for (int j = 0; j < 8; ++j) {
    dq[j] = *(const unsigned*)&sT[wi][c0 + 2 * j];
    dr[j] = *(const unsigned*)&sR[wi][c0 + 2 * j];
    dc[j] = *(const unsigned*)&sW[wi][c0 + 2 * j];
  }
}

// ---------------------------------------------------------------------------
// Axial attention over one line (B*128 workgroups, 16 waves, 2 waves/head).
// mode 0 (rows): writes fused^T bf16 into outT [b][h][w][c] (query of pass 2)
// mode 1 (cols): writes relu(fused + tgt) f32 into outF [b][c][h][w]
// ---------------------------------------------------------------------------
__global__ __launch_bounds__(NTHR, 1) void axial_attn_kernel(
    const bf16_t* __restrict__ Tq, const bf16_t* __restrict__ Tkv,
    const bf16_t* __restrict__ WqT, const bf16_t* __restrict__ WkT,
    const bf16_t* __restrict__ WvT, const bf16_t* __restrict__ WoT,
    const float*  __restrict__ bias,          // [8][255]
    bf16_t* __restrict__ outT, float* __restrict__ outF,
    const float* __restrict__ tgt, int mode)
{
  extern __shared__ char smem[];
  bf16_t* sX = (bf16_t*)(smem + OFF_X);   // X^T [l][c]   (Xq then Xkv)
  bf16_t* sQ = (bf16_t*)(smem + OFF_Q);   // Q^T [q][c]
  bf16_t* sK = (bf16_t*)(smem + OFF_K);   // K^T [k][c]
  bf16_t* sV = (bf16_t*)(smem + OFF_V);   // V   [d][l]
  bf16_t* sP = (bf16_t*)(smem + OFF_P);   // per-wave P bounce
  float*  sB = (float*)(smem + OFF_B);    // bias rows
  bf16_t* sO = sX;                        // attn-out^T [q][c], aliases sX
  float*  sF = (float*)(smem + OFF_Q);    // mode-1 fused^T f32, aliases sQ+sK
  bf16_t* sF16 = (bf16_t*)(smem + OFF_Q); // mode-0 fused^T bf16, aliases sQ

  const int tid  = threadIdx.x;
  const int lane = tid & 31;
  const int wave = tid >> 5;
  const int kh   = lane >> 4, ln = lane & 15;
  const int b    = blockIdx.x >> 7;
  const int line = blockIdx.x & 127;
  const size_t slice = ((size_t)(b * LSEQ + line)) << 15;   // *128*256

  // ---- phase 1: Xq -> sX (async DMA to LDS), bias -> sB -------------------
  {
    const bf16_t* g = Tq + slice;
    #pragma unroll
    for (int i = 0; i < 8; ++i) {
      int idx = (i << 9) + tid;              // 4096 x 16B chunks
      int row = idx >> 4, c16 = idx & 15;
      async_ld_b128((unsigned)(OFF_X + (row * PX + c16 * 8) * 2),
                    (const void*)(g + row * 256 + c16 * 8));
    }
    for (int i = tid; i < NHEAD * 255; i += NTHR)
      sB[(i / 255) * 256 + (i % 255)] = bias[i];
    wait_async0();
  }
  __syncthreads();

  // ---- phase 2: Q = Wq^T X, stored transposed [q][c] ----------------------
  for (int t = wave; t < 128; t += NWAVE) {
    const int dt = t >> 3, lt = t & 7;
    AccTile acc;
    #pragma unroll
    for (int r = 0; r < 8; ++r) acc.f[r] = 0.f;
    #pragma unroll
    for (int kk = 0; kk < 8; ++kk) {
      v16bf a  = load_a_tile(WqT + dt * 16 * 256 + kk * 32, 256, lane);
      v16bf bb = load_b_tile(sX + lt * 16 * PX + kk * 32, PX, lane);
      acc.v = wmma_bf16(a, bb, acc.v);
    }
    const int n  = lt * 16 + ln;
    const int m0 = dt * 16 + (kh << 3);
    unsigned* p = (unsigned*)(sQ + n * PX + m0);
    #pragma unroll
    for (int r = 0; r < 8; r += 2) p[r >> 1] = pack2bf(acc.f[r], acc.f[r + 1]);
  }
  __syncthreads();

  // ---- phase 3: Xkv -> sX (async DMA to LDS) ------------------------------
  {
    const bf16_t* g = Tkv + slice;
    #pragma unroll
    for (int i = 0; i < 8; ++i) {
      int idx = (i << 9) + tid;
      int row = idx >> 4, c16 = idx & 15;
      async_ld_b128((unsigned)(OFF_X + (row * PX + c16 * 8) * 2),
                    (const void*)(g + row * 256 + c16 * 8));
    }
    wait_async0();
  }
  __syncthreads();

  // ---- phase 4: K -> sK (transposed), V -> sV [d][l] ----------------------
  for (int t = wave; t < 128; t += NWAVE) {
    const int dt = t >> 3, lt = t & 7;
    AccTile acc;
    #pragma unroll
    for (int r = 0; r < 8; ++r) acc.f[r] = 0.f;
    #pragma unroll
    for (int kk = 0; kk < 8; ++kk) {
      v16bf a  = load_a_tile(WkT + dt * 16 * 256 + kk * 32, 256, lane);
      v16bf bb = load_b_tile(sX + lt * 16 * PX + kk * 32, PX, lane);
      acc.v = wmma_bf16(a, bb, acc.v);
    }
    const int n  = lt * 16 + ln;
    const int m0 = dt * 16 + (kh << 3);
    unsigned* p = (unsigned*)(sK + n * PX + m0);
    #pragma unroll
    for (int r = 0; r < 8; r += 2) p[r >> 1] = pack2bf(acc.f[r], acc.f[r + 1]);
  }
  for (int t = wave; t < 128; t += NWAVE) {
    const int dt = t >> 3, lt = t & 7;
    AccTile acc;
    #pragma unroll
    for (int r = 0; r < 8; ++r) acc.f[r] = 0.f;
    #pragma unroll
    for (int kk = 0; kk < 8; ++kk) {
      v16bf a  = load_a_tile(WvT + dt * 16 * 256 + kk * 32, 256, lane);
      v16bf bb = load_b_tile(sX + lt * 16 * PX + kk * 32, PX, lane);
      acc.v = wmma_bf16(a, bb, acc.v);
    }
    const int n  = lt * 16 + ln;
    const int m0 = dt * 16 + (kh << 3);
    #pragma unroll
    for (int r = 0; r < 8; ++r) sV[(m0 + r) * PVV + n] = f2bf(acc.f[r]);
  }
  __syncthreads();

  // ---- phase 5: flash attention; head = wave/2, q-blocks split by parity --
  {
    const int h = wave >> 1;
    const float scale = 0.17677669529663687f;   // 1/sqrt(32)
    bf16_t* myP = sP + wave * (16 * PP);
    for (int qb = (wave & 1); qb < 8; qb += 2) {
      v16bf aQ = load_a_tile(sQ + qb * 16 * PX + h * DH, PX, lane);
      float mrun[8], lrun[8];
      AccTile o0, o1;
      #pragma unroll
      for (int r = 0; r < 8; ++r) { mrun[r] = -3.0e38f; lrun[r] = 0.f; o0.f[r] = 0.f; o1.f[r] = 0.f; }
      for (int kc = 0; kc < 4; ++kc) {
        AccTile S0, S1;
        {
          v8f z = {0.f,0.f,0.f,0.f,0.f,0.f,0.f,0.f};
          v16bf bK0 = load_b_tile(sK + (kc * 32) * PX + h * DH, PX, lane);
          v16bf bK1 = load_b_tile(sK + (kc * 32 + 16) * PX + h * DH, PX, lane);
          S0.v = wmma_bf16(aQ, bK0, z);
          S1.v = wmma_bf16(aQ, bK1, z);
        }
        const int kpos = kc * 32 + ln;
        #pragma unroll
        for (int r = 0; r < 8; ++r) {
          int q = qb * 16 + (kh << 3) + r;
          S0.f[r] = S0.f[r] * scale + sB[h * 256 + (q - kpos + 127)];
          S1.f[r] = S1.f[r] * scale + sB[h * 256 + (q - kpos - 16 + 127)];
        }
        #pragma unroll
        for (int r = 0; r < 8; ++r) {
          float mx = fmaxf(S0.f[r], S1.f[r]);
          mx = fmaxf(mx, __shfl_xor(mx, 1, 32));
          mx = fmaxf(mx, __shfl_xor(mx, 2, 32));
          mx = fmaxf(mx, __shfl_xor(mx, 4, 32));
          mx = fmaxf(mx, __shfl_xor(mx, 8, 32));
          float mnew  = fmaxf(mrun[r], mx);
          float alpha = __expf(mrun[r] - mnew);
          float p0 = __expf(S0.f[r] - mnew);
          float p1 = __expf(S1.f[r] - mnew);
          S0.f[r] = p0; S1.f[r] = p1;
          float rs = p0 + p1;
          rs += __shfl_xor(rs, 1, 32);
          rs += __shfl_xor(rs, 2, 32);
          rs += __shfl_xor(rs, 4, 32);
          rs += __shfl_xor(rs, 8, 32);
          lrun[r] = lrun[r] * alpha + rs;
          mrun[r] = mnew;
          o0.f[r] *= alpha;
          o1.f[r] *= alpha;
        }
        #pragma unroll
        for (int r = 0; r < 8; ++r) {
          int rq = (kh << 3) + r;
          myP[rq * PP + ln]      = f2bf(S0.f[r]);
          myP[rq * PP + 16 + ln] = f2bf(S1.f[r]);
        }
        __asm__ volatile("s_wait_dscnt 0" ::: "memory");
        v16bf aP  = load_a_tile(myP, PP, lane);
        v16bf bV0 = load_b_tile(sV + (h * DH) * PVV + kc * 32, PVV, lane);
        v16bf bV1 = load_b_tile(sV + (h * DH + 16) * PVV + kc * 32, PVV, lane);
        o0.v = wmma_bf16(aP, bV0, o0.v);
        o1.v = wmma_bf16(aP, bV1, o1.v);
      }
      #pragma unroll
      for (int r = 0; r < 8; ++r) {
        int q = qb * 16 + (kh << 3) + r;
        float inv = 1.0f / lrun[r];
        sO[q * PX + h * DH + ln]      = f2bf(o0.f[r] * inv);
        sO[q * PX + h * DH + 16 + ln] = f2bf(o1.f[r] * inv);
      }
    }
  }
  __syncthreads();

  // ---- phase 6: output projection (Wo); stage into LDS --------------------
  for (int t = wave; t < 128; t += NWAVE) {
    const int lt = t >> 4, dt = t & 15;
    AccTile acc;
    #pragma unroll
    for (int r = 0; r < 8; ++r) acc.f[r] = 0.f;
    #pragma unroll
    for (int kk = 0; kk < 8; ++kk) {
      v16bf a  = load_a_tile(sO + lt * 16 * PX + kk * 32, PX, lane);
      v16bf bb = load_b_tile(WoT + dt * 16 * 256 + kk * 32, 256, lane);
      acc.v = wmma_bf16(a, bb, acc.v);
    }
    const int n  = dt * 16 + ln;
    const int m0 = lt * 16 + (kh << 3);
    if (mode == 0) {
      #pragma unroll
      for (int r = 0; r < 8; ++r) sF16[(m0 + r) * PX + n] = f2bf(acc.f[r]);
    } else {
      #pragma unroll
      for (int r = 0; r < 8; ++r) sF[(m0 + r) * PF + n] = acc.f[r];
    }
  }
  __syncthreads();

  if (mode == 0) {
    // ---- phase 7a: coalesced bf16 rows -> outT [b][h=l][w=line][c] --------
    #pragma unroll
    for (int i = 0; i < 8; ++i) {
      int idx = (i << 9) + tid;
      int row = idx >> 4, c16 = idx & 15;
      *(uint4*)(outT + (((size_t)(b * LSEQ + row)) * LSEQ + line) * CCH + c16 * 8) =
          *(const uint4*)(sF16 + row * PX + c16 * 8);
    }
  } else {
    // ---- phase 7b: relu(fused + tgt) -> d_out, coalesced on w -------------
    const int w = tid & 127;
    for (int c = tid >> 7; c < CCH; c += 4) {
      size_t gi = (((size_t)b * CCH + c) * LSEQ + line) * LSEQ + w;
      float v = sF[w * PF + c] + tgt[gi];
      outF[gi] = fmaxf(v, 0.f);
    }
  }
}

// ---------------------------------------------------------------------------
// Host launcher
// ---------------------------------------------------------------------------
extern "C" void kernel_launch(void* const* d_in, const int* in_sizes, int n_in,
                              void* d_out, int out_size, void* d_ws, size_t ws_size,
                              hipStream_t stream) {
  (void)in_sizes; (void)n_in; (void)out_size; (void)ws_size;
  const float* tgt = (const float*)d_in[0];
  const float* ref = (const float*)d_in[1];
  float* out = (float*)d_out;
  float* ref_out = out + (size_t)BATCH * CCH * LSEQ * LSEQ;   // tuple second element

  char* ws = (char*)d_ws;
  const size_t TSZ = (size_t)BATCH * LSEQ * LSEQ * CCH;       // 16,777,216 elems
  bf16_t* WT      = (bf16_t*)ws;                               // 8 * 65536 bf16
  bf16_t* Tq_row  = (bf16_t*)(ws + (1u << 20));
  bf16_t* Tkv_row = Tq_row + TSZ;
  bf16_t* Tkv_col = Tkv_row + TSZ;
  bf16_t* Tq_col  = Tkv_col + TSZ;

  for (int m = 0; m < 8; ++m)
    pack_w_kernel<<<256, 256, 0, stream>>>((const float*)d_in[10 + m],
                                           WT + (size_t)m * 65536);

  prep_kernel<<<dim3(8, 128, 4), 256, 0, stream>>>(
      tgt, ref,
      (const float*)d_in[2], (const float*)d_in[3],
      (const float*)d_in[4], (const float*)d_in[5],
      (const float*)d_in[6], (const float*)d_in[7],
      (const float*)d_in[8], (const float*)d_in[9],
      Tq_row, Tkv_row, Tkv_col, ref_out);

  hipFuncSetAttribute((const void*)axial_attn_kernel,
                      hipFuncAttributeMaxDynamicSharedMemorySize, SMEM_ATT);

  // pass 1: rows (along H); K/V from bn(ref); writes query tensor for pass 2
  axial_attn_kernel<<<512, NTHR, SMEM_ATT, stream>>>(
      Tq_row, Tkv_row,
      WT + 0 * 65536, WT + 1 * 65536, WT + 2 * 65536, WT + 3 * 65536,
      (const float*)d_in[18], Tq_col, (float*)nullptr, (const float*)nullptr, 0);

  // pass 2: cols (along W); K/V from raw ref; fused epilogue relu(+tgt)
  axial_attn_kernel<<<512, NTHR, SMEM_ATT, stream>>>(
      Tq_col, Tkv_col,
      WT + 4 * 65536, WT + 5 * 65536, WT + 6 * 65536, WT + 7 * 65536,
      (const float*)d_in[19], (bf16_t*)nullptr, out, tgt, 1);
}
